// Repro_18897856102677
// MI455X (gfx1250) — matrix operand order repaired and checked by reading, not compiled, not run on hardware
//
#include <hip/hip_runtime.h>
#include <hip/hip_bf16.h>

// ---------------------------------------------------------------------------
// Transformer encoder layer for MI455X (gfx1250, wave32, WMMA 16x16x32 f16).
// B=8, S=1024, D=512, H=16, HD=32, FF=2048.  Compute-bound -> everything
// routed through v_wmma_f32_16x16x32_f16; tile staging via the Tensor Data
// Mover (tensor_load_to_lds + s_wait_tensorcnt) where no transpose is needed.
// ---------------------------------------------------------------------------

typedef _Float16 h16;
typedef __attribute__((ext_vector_type(4)))  _Float16 v4h;
typedef __attribute__((ext_vector_type(8)))  _Float16 v8h;
typedef __attribute__((ext_vector_type(16))) _Float16 v16h;
typedef __attribute__((ext_vector_type(8)))  float    v8f;
typedef __attribute__((ext_vector_type(4)))  unsigned int u32x4;
typedef __attribute__((ext_vector_type(8)))  int      i32x8;
typedef __attribute__((ext_vector_type(4)))  int      i32x4;

union AFrag { v16h v; v8h h[2]; };

#define TB   8
#define TS   1024
#define TD   512
#define TH   16
#define THD  32
#define TFF  2048
#define TM   (TB * TS)          // 8192 rows
#define INV_SCALE 0.17677669529663687f   // 1/sqrt(32)
#define LDP  40                  // padded LDS pitch (halfs): conflict-free

__device__ __forceinline__ float wave16_max(float v) {
    #pragma unroll
    for (int m = 1; m < 16; m <<= 1) v = fmaxf(v, __shfl_xor(v, m, 32));
    return v;
}
__device__ __forceinline__ float wave16_sum(float v) {
    #pragma unroll
    for (int m = 1; m < 16; m <<= 1) v += __shfl_xor(v, m, 32);
    return v;
}
__device__ __forceinline__ float wave32_sum(float v) {
    #pragma unroll
    for (int m = 1; m < 32; m <<= 1) v += __shfl_xor(v, m, 32);
    return v;
}

// ---------------------------------------------------------------------------
// TDM: async 2D tile load global -> LDS with hardware row padding.
// Tile: tile_d0 elements (h16) per row, tile_d1 rows, global row stride
// `stride` (elements).  LDS padding: +4 DWORDs every 16 DWORDs, i.e. rows of
// 32 halfs land on a 40-half pitch (matches LDP).
// Descriptor packing per CDNA5 ISA 8.3/8.4 (D# group0/group1).
// Toolchain here uses the 6-arg builtin form:
//   (u32x4 g0, i32x8 g1, i32x4 g2, i32x4 g3, i32x8 g4, i32 cpol)
// ---------------------------------------------------------------------------
__device__ __forceinline__ void tdm_load_tile_2d(const h16* gsrc, void* lds_dst,
                                                 int tile_d0, int tile_d1,
                                                 int dim0, int dim1, int stride) {
    unsigned long long ga = (unsigned long long)(size_t)gsrc;
    unsigned lds = (unsigned)(size_t)lds_dst;
    u32x4 g0;
    g0[0] = 1u;                                            // count=1, user mode
    g0[1] = lds;                                           // lds_addr
    g0[2] = (unsigned)ga;                                  // global_addr[31:0]
    g0[3] = (unsigned)((ga >> 32) & 0x1FFFFFFull)          // global_addr[56:32]
          | (2u << 30);                                    // type = 2 (image)
    i32x8 g1;
    g1[0] = (1 << 16)                                      // data_size = 1 -> 2B
          | (1 << 20)                                      // pad_enable
          | (3 << 22)                                      // pad_interval: 16 DW
          | (3 << 25);                                     // pad_amount: 4 DW
    g1[1] = (int)(((unsigned)dim0 & 0xFFFFu) << 16);       // tensor_dim0 lo16
    g1[2] = (int)(((unsigned)dim0 >> 16)
          | (((unsigned)dim1 & 0xFFFFu) << 16));           // dim0 hi, dim1 lo
    g1[3] = (int)(((unsigned)dim1 >> 16)
          | (((unsigned)tile_d0 & 0xFFFFu) << 16));        // dim1 hi, tile_dim0
    g1[4] = (int)((unsigned)tile_d1 & 0xFFFFu);            // tile_dim1, tile_dim2=0
    g1[5] = stride;                                        // tensor_dim0_stride lo32
    g1[6] = 0;
    g1[7] = 0;
    i32x4 gz4;
    gz4[0] = 0; gz4[1] = 0; gz4[2] = 0; gz4[3] = 0;
    i32x8 gz8;
    gz8[0] = 0; gz8[1] = 0; gz8[2] = 0; gz8[3] = 0;
    gz8[4] = 0; gz8[5] = 0; gz8[6] = 0; gz8[7] = 0;
    __builtin_amdgcn_tensor_load_to_lds(g0, g1, gz4, gz4, gz8, 0);
}

// ---------------------------------------------------------------------------
// Generic WMMA GEMM:  C[M,N] = A[M,K] * W[K,N] + bias (+ residual | gelu)
// Block: 256 threads (8 waves), tile 128x128, BK=32.
// A tile staged by the TDM; B tile staged manually (needs transpose).
// EPI: 0 = bias, 1 = bias + residual, 2 = bias + exact-erf GELU
// ---------------------------------------------------------------------------
template <int EPI>
__global__ __launch_bounds__(256)
void gemm_wmma_f16(const h16* __restrict__ A, const h16* __restrict__ W,
                   const h16* __restrict__ bias, const h16* __restrict__ res,
                   h16* __restrict__ C, int M, int N, int K) {
    __shared__ alignas(16) h16 As[128 * LDP];
    __shared__ alignas(16) h16 Bs[128 * LDP];

    const int tid  = threadIdx.x;
    const int lane = tid & 31;
    const int wave = tid >> 5;
    const int lr   = lane & 15;
    const int half = lane >> 4;
    const int wm   = wave >> 2;      // 0..1
    const int wn   = wave & 3;       // 0..3
    const int m0   = blockIdx.y * 128;
    const int n0   = blockIdx.x * 128;

    v8f acc[4][2];
    #pragma unroll
    for (int i = 0; i < 4; ++i)
        #pragma unroll
        for (int j = 0; j < 2; ++j)
            acc[i][j] = (v8f)0.0f;

    const int bkk  = tid >> 3;            // 0..31
    const int bng  = (tid & 7) * 16;      // n sub-range 0..112

    for (int k0 = 0; k0 < K; k0 += 32) {
        // ---- stage A tile via Tensor Data Mover (async DMA + HW padding) ----
        if (wave == 0) {
            tdm_load_tile_2d(A + (size_t)m0 * K + k0, As,
                             /*tile*/32, 128, /*tensor*/K, M, /*stride*/K);
        }
        // ---- stage B tile transposed: Bs[n][k] ----
        {
            const h16* src = W + (size_t)(k0 + bkk) * N + n0 + bng;
            v8h t0 = *(const v8h*)(src);
            v8h t1 = *(const v8h*)(src + 8);
            #pragma unroll
            for (int j = 0; j < 8; ++j) {
                Bs[(bng + j)     * LDP + bkk] = t0[j];
                Bs[(bng + 8 + j) * LDP + bkk] = t1[j];
            }
            if (k0 + 32 < K)
                __builtin_prefetch((const void*)(src + 32 * (size_t)N), 0, 1);
        }
        if (wave == 0)
            __builtin_amdgcn_s_wait_tensorcnt(0);
        __syncthreads();

        // ---- fragments + WMMA ----
        AFrag afr[4];
        #pragma unroll
        for (int i = 0; i < 4; ++i) {
            const h16* p = As + (wm * 64 + i * 16 + lr) * LDP;
            afr[i].h[0] = *(const v8h*)(p + 8 * half);        // K 0..7 / 8..15
            afr[i].h[1] = *(const v8h*)(p + 16 + 8 * half);   // K 16..23 / 24..31
        }
        AFrag bfr[2];
        #pragma unroll
        for (int j = 0; j < 2; ++j) {
            const h16* p = Bs + (wn * 32 + j * 16 + lr) * LDP + 8 * half;
            bfr[j].h[0] = *(const v8h*)(p);
            bfr[j].h[1] = *(const v8h*)(p + 16);
        }
        #pragma unroll
        for (int i = 0; i < 4; ++i)
            #pragma unroll
            for (int j = 0; j < 2; ++j)
                acc[i][j] = __builtin_amdgcn_wmma_f32_16x16x32_f16(
                    false, afr[i].v, false, bfr[j].v,
                    (short)0, acc[i][j], false, false);
        __syncthreads();
    }

    // ---- epilogue ----
    #pragma unroll
    for (int i = 0; i < 4; ++i) {
        #pragma unroll
        for (int j = 0; j < 2; ++j) {
            const int col = n0 + wn * 32 + j * 16 + lr;
            const float bv = (float)bias[col];
            #pragma unroll
            for (int r = 0; r < 8; ++r) {
                const int row = m0 + wm * 64 + i * 16 + r + 8 * half;
                float v = acc[i][j][r] + bv;
                if (EPI == 1) v += (float)res[(size_t)row * N + col];
                if (EPI == 2) v = 0.5f * v * (1.0f + erff(v * 0.70710678118654752f));
                C[(size_t)row * N + col] = (h16)v;
            }
        }
    }
}

// ---------------------------------------------------------------------------
// Attention: flash-style, one block per (b, h, 128-row q-block).
// 8 waves; each wave owns 16 q rows.  K chunk staged by TDM, V staged
// manually transposed.  Online softmax; P round-trips per-wave LDS to convert
// C-layout -> A-layout for the P*V WMMA.
// ---------------------------------------------------------------------------
__global__ __launch_bounds__(256)
void attention_wmma(const h16* __restrict__ qkv, const h16* __restrict__ mask,
                    h16* __restrict__ ctx_out) {
    __shared__ alignas(16) h16 Kl[32 * LDP];       // [kcol][d]
    __shared__ alignas(16) h16 Vl[32 * LDP];       // [d][krow]  (transposed)
    __shared__ alignas(16) h16 Pl[8 * 16 * LDP];   // per-wave [16][32]

    const int tid  = threadIdx.x;
    const int lane = tid & 31;
    const int wave = tid >> 5;
    const int lr   = lane & 15;
    const int half = lane >> 4;
    const int b    = blockIdx.z;
    const int h    = blockIdx.y;
    const int q0   = blockIdx.x * 128;
    const int qrow = q0 + wave * 16;

    const size_t rowpitch = 3 * TD;                 // qkv row stride in halfs
    const h16* qbase = qkv + h * THD;               // Q at +0
    const h16* kbase = qkv + TD + h * THD;          // K at +D
    const h16* vbase = qkv + 2 * TD + h * THD;      // V at +2D

    // Q fragment (A-layout, 16 rows x 32 d) held in registers for all chunks
    AFrag qf;
    {
        const h16* p = qbase + (size_t)(b * TS + qrow + lr) * rowpitch;
        qf.h[0] = *(const v8h*)(p + 8 * half);
        qf.h[1] = *(const v8h*)(p + 16 + 8 * half);
    }

    float mrow[8], lrow[8];
    #pragma unroll
    for (int r = 0; r < 8; ++r) { mrow[r] = -1e30f; lrow[r] = 0.0f; }
    v8f ctx0 = (v8f)0.0f, ctx1 = (v8f)0.0f;

    const int skrow = tid >> 3;          // 0..31
    const int sd0   = (tid & 7) * 4;     // 0..28 (4 halfs each)

    for (int kc = 0; kc < TS; kc += 32) {
        // ---- K chunk via TDM (row-major, HW-padded to LDP pitch) ----
        if (wave == 0) {
            tdm_load_tile_2d(kbase + (size_t)(b * TS + kc) * rowpitch, Kl,
                             /*tile*/32, 32, /*tensor*/(int)rowpitch, TB * TS,
                             /*stride*/(int)rowpitch);
        }
        // ---- V chunk staged manually, transposed ----
        {
            const h16* vs = vbase + (size_t)(b * TS + kc + skrow) * rowpitch + sd0;
            v4h vv = *(const v4h*)(vs);
            #pragma unroll
            for (int jj = 0; jj < 4; ++jj)
                Vl[(sd0 + jj) * LDP + skrow] = vv[jj];
            if (kc + 32 < TS)
                __builtin_prefetch((const void*)(vs + 32 * rowpitch), 0, 1);
        }
        if (wave == 0)
            __builtin_amdgcn_s_wait_tensorcnt(0);
        __syncthreads();

        // ---- scores: two 16-col tiles, each one wmma (K-dim = HD = 32) ----
        v8f s[2];
        #pragma unroll
        for (int j = 0; j < 2; ++j) {
            AFrag kf;
            const h16* p = Kl + (j * 16 + lr) * LDP + 8 * half;
            kf.h[0] = *(const v8h*)(p);
            kf.h[1] = *(const v8h*)(p + 16);
            s[j] = __builtin_amdgcn_wmma_f32_16x16x32_f16(
                false, qf.v, false, kf.v, (short)0, (v8f)0.0f, false, false);
        }

        // ---- masked logits + online softmax ----
        float mb[2];
        #pragma unroll
        for (int j = 0; j < 2; ++j) {
            const float mv = (float)mask[(size_t)b * TS + kc + j * 16 + lr];
            mb[j] = (1.0f - mv) * 10000.0f;
        }
        h16* pw = Pl + wave * 16 * LDP;
        #pragma unroll
        for (int r = 0; r < 8; ++r) {
            float l0 = s[0][r] * INV_SCALE - mb[0];
            float l1 = s[1][r] * INV_SCALE - mb[1];
            float cm = wave16_max(fmaxf(l0, l1));
            float mn = fmaxf(mrow[r], cm);
            float sc = expf(mrow[r] - mn);
            mrow[r]  = mn;
            float p0 = expf(l0 - mn);
            float p1 = expf(l1 - mn);
            lrow[r]  = lrow[r] * sc + wave16_sum(p0 + p1);
            ctx0[r] *= sc;
            ctx1[r] *= sc;
            pw[(r + 8 * half) * LDP + lr]      = (h16)p0;
            pw[(r + 8 * half) * LDP + 16 + lr] = (h16)p1;
        }

        // ---- P (A-layout reload) * V chunk ----
        AFrag pf;
        {
            const h16* p = pw + lr * LDP;
            pf.h[0] = *(const v8h*)(p + 8 * half);
            pf.h[1] = *(const v8h*)(p + 16 + 8 * half);
        }
        #pragma unroll
        for (int j = 0; j < 2; ++j) {
            AFrag vf;
            const h16* p = Vl + (j * 16 + lr) * LDP + 8 * half;
            vf.h[0] = *(const v8h*)(p);
            vf.h[1] = *(const v8h*)(p + 16);
            v8f& cc = (j == 0) ? ctx0 : ctx1;
            cc = __builtin_amdgcn_wmma_f32_16x16x32_f16(
                false, pf.v, false, vf.v, (short)0, cc, false, false);
        }
        __syncthreads();
    }

    // ---- normalize + write ctx [B,S,D] with d = h*32 + col ----
    #pragma unroll
    for (int r = 0; r < 8; ++r) {
        const float inv = 1.0f / lrow[r];
        const int   row = qrow + r + 8 * half;
        h16* o = ctx_out + (size_t)(b * TS + row) * TD + h * THD;
        o[lr]      = (h16)(ctx0[r] * inv);
        o[16 + lr] = (h16)(ctx1[r] * inv);
    }
}

// ---------------------------------------------------------------------------
// LayerNorm over D=512: one wave per row (16 f16 per lane), wave32 shuffles.
// ---------------------------------------------------------------------------
__global__ __launch_bounds__(256)
void layernorm_f16(const h16* __restrict__ x, const h16* __restrict__ w,
                   const h16* __restrict__ bta, h16* __restrict__ y, int rows) {
    const int lane = threadIdx.x & 31;
    const int wave = threadIdx.x >> 5;
    const int row  = blockIdx.x * 8 + wave;
    if (row >= rows) return;

    const h16* p = x + (size_t)row * TD + lane * 16;
    v8h a = *(const v8h*)(p);
    v8h bvec = *(const v8h*)(p + 8);

    float s = 0.0f, ss = 0.0f;
    float xv[16];
    #pragma unroll
    for (int i = 0; i < 8; ++i) { xv[i] = (float)a[i]; xv[8 + i] = (float)bvec[i]; }
    #pragma unroll
    for (int i = 0; i < 16; ++i) { s += xv[i]; ss += xv[i] * xv[i]; }
    s  = wave32_sum(s);
    ss = wave32_sum(ss);
    const float mu  = s * (1.0f / TD);
    const float var = ss * (1.0f / TD) - mu * mu;
    const float rs  = rsqrtf(var + 1e-5f);

    h16* o = y + (size_t)row * TD + lane * 16;
    const h16* wp = w + lane * 16;
    const h16* bp = bta + lane * 16;
    #pragma unroll
    for (int i = 0; i < 16; ++i)
        o[i] = (h16)((xv[i] - mu) * rs * (float)wp[i] + (float)bp[i]);
}

// ---------------------------------------------------------------------------
// Launch: QKV gemm -> attention -> out-proj(+res) -> LN1 -> FF1(+gelu)
//         -> FF2(+res) -> LN2
// ---------------------------------------------------------------------------
extern "C" void kernel_launch(void* const* d_in, const int* in_sizes, int n_in,
                              void* d_out, int out_size, void* d_ws, size_t ws_size,
                              hipStream_t stream) {
    (void)in_sizes; (void)n_in; (void)out_size; (void)ws_size;
    const h16* x     = (const h16*)d_in[0];
    const h16* w_qkv = (const h16*)d_in[1];
    const h16* b_qkv = (const h16*)d_in[2];
    const h16* mask  = (const h16*)d_in[3];
    const h16* w_o   = (const h16*)d_in[4];
    const h16* b_o   = (const h16*)d_in[5];
    const h16* ln1_b = (const h16*)d_in[6];
    const h16* ln1_w = (const h16*)d_in[7];
    const h16* w_ff1 = (const h16*)d_in[8];
    const h16* b_ff1 = (const h16*)d_in[9];
    const h16* w_ff2 = (const h16*)d_in[10];
    const h16* b_ff2 = (const h16*)d_in[11];
    const h16* ln2_b = (const h16*)d_in[12];
    const h16* ln2_w = (const h16*)d_in[13];

    char* ws = (char*)d_ws;
    h16* qkv = (h16*)(ws);                                     // 8192*1536
    h16* ctx = (h16*)(ws + (size_t)TM * 3 * TD * 2);           // 8192*512
    h16* hb  = (h16*)(ws + (size_t)TM * 3 * TD * 2 + (size_t)TM * TD * 2);
    h16* x1  = (h16*)(ws + (size_t)TM * 3 * TD * 2 + (size_t)TM * TD * 4);
    h16* ff  = (h16*)(ws + (size_t)TM * 3 * TD * 2 + (size_t)TM * TD * 6);
    h16* h2  = (h16*)(ws + (size_t)TM * 3 * TD * 2 + (size_t)TM * TD * 6
                         + (size_t)TM * TFF * 2);

    dim3 blk(256);

    // 1) QKV projection: [8192,512] x [512,1536]
    gemm_wmma_f16<0><<<dim3(3 * TD / 128, TM / 128), blk, 0, stream>>>(
        x, w_qkv, b_qkv, nullptr, qkv, TM, 3 * TD, TD);

    // 2) Attention
    attention_wmma<<<dim3(TS / 128, TH, TB), blk, 0, stream>>>(qkv, mask, ctx);

    // 3) Output projection + residual
    gemm_wmma_f16<1><<<dim3(TD / 128, TM / 128), blk, 0, stream>>>(
        ctx, w_o, b_o, x, hb, TM, TD, TD);

    // 4) LayerNorm 1
    layernorm_f16<<<dim3(TM / 8), blk, 0, stream>>>(hb, ln1_w, ln1_b, x1, TM);

    // 5) FFN up + GELU: [8192,512] x [512,2048]
    gemm_wmma_f16<2><<<dim3(TFF / 128, TM / 128), blk, 0, stream>>>(
        x1, w_ff1, b_ff1, nullptr, ff, TM, TFF, TD);

    // 6) FFN down + residual: [8192,2048] x [2048,512]
    gemm_wmma_f16<1><<<dim3(TD / 128, TM / 128), blk, 0, stream>>>(
        ff, w_ff2, b_ff2, x1, h2, TM, TD, TFF);

    // 7) LayerNorm 2 -> output (f16)
    layernorm_f16<<<dim3(TM / 8), blk, 0, stream>>>(h2, ln2_w, ln2_b, (h16*)d_out, TM);
}
